// DeepSeekMoELayer_37297495998514
// MI455X (gfx1250) — matrix operand order repaired and checked by reading, not compiled, not run on hardware
//
#include <hip/hip_runtime.h>
#include <hip/hip_bf16.h>
#include <math.h>

// ---------------------------------------------------------------------------
// Types for WMMA fragments (wave32, gfx1250)
// ---------------------------------------------------------------------------
typedef __attribute__((ext_vector_type(16))) __bf16       bf16x16;
typedef __attribute__((ext_vector_type(8)))  float        f32x8;
typedef __attribute__((ext_vector_type(4)))  unsigned int u32x4;

union FragU { u32x4 u[2]; bf16x16 v; };

static __device__ __forceinline__ unsigned short f32_to_bf16(float x) {
  unsigned int u = __float_as_uint(x);
  u += 0x7FFFu + ((u >> 16) & 1u);      // round-to-nearest-even
  return (unsigned short)(u >> 16);
}

// A fragment: 16x32 bf16, M rows striped over lanes 0..15 / 16..31.
// lane L (<16): M=L, K = k0+0..7 and k0+16..23 ; lane L (>=16): M=L-16, K offset +8.
static __device__ __forceinline__ bf16x16
load_a_frag(const unsigned short* A, int K, int row, int k0, int lane) {
  const unsigned short* p =
      A + (size_t)(row + (lane & 15)) * K + (k0 + ((lane >> 4) << 3));
  FragU f;
  f.u[0] = *(const u32x4*)(p);
  f.u[1] = *(const u32x4*)(p + 16);
  return f.v;
}

// B fragment: 32x16 bf16, N cols striped over lanes; lanes 0..15 hold K=k0+0..7 and k0+16..23,
// lanes 16..31 hold K offset +8 (same order as the A fragment).
static __device__ __forceinline__ bf16x16
load_b_frag(const unsigned short* B, int K, int col, int k0, int lane) {
  const unsigned short* p =
      B + (size_t)(col + (lane & 15)) * K + (k0 + ((lane >> 4) << 3));
  FragU f;
  f.u[0] = *(const u32x4*)(p);
  f.u[1] = *(const u32x4*)(p + 16);
  return f.v;
}

static __device__ __forceinline__ f32x8 wmma_bf16(bf16x16 a, bf16x16 b, f32x8 c) {
  return __builtin_amdgcn_wmma_f32_16x16x32_bf16(
      /*neg_a=*/false, a, /*neg_b=*/false, b,
      /*c_mod=*/(short)0, c, /*reuse_a=*/false, /*reuse_b=*/false);
}

// Accumulate a 64(M) x 32(N) wave tile: D += A[m0.., :K] * B[n0.., :K]^T
// (both operands row-major [rows, K], K contiguous).
static __device__ __forceinline__ void
mma_accum(const unsigned short* __restrict__ A, const unsigned short* __restrict__ B,
          int K, int m0, int n0, int lane, f32x8 acc[4][2]) {
  for (int k0 = 0; k0 < K; k0 += 32) {
    if (k0 + 32 < K) {
      __builtin_prefetch(A + (size_t)(m0 + (lane & 15)) * K + k0 + 32, 0, 0);
      __builtin_prefetch(B + (size_t)(n0 + (lane & 15)) * K + k0 + 32, 0, 0);
    }
    bf16x16 a[4], b[2];
#pragma unroll
    for (int mi = 0; mi < 4; ++mi) a[mi] = load_a_frag(A, K, m0 + mi * 16, k0, lane);
#pragma unroll
    for (int ni = 0; ni < 2; ++ni) b[ni] = load_b_frag(B, K, n0 + ni * 16, k0, lane);
#pragma unroll
    for (int mi = 0; mi < 4; ++mi)
#pragma unroll
      for (int ni = 0; ni < 2; ++ni)
        acc[mi][ni] = wmma_bf16(a[mi], b[ni], acc[mi][ni]);
  }
}

// ---------------------------------------------------------------------------
// fp32 -> bf16 conversion
// ---------------------------------------------------------------------------
__global__ void __launch_bounds__(256)
cvt_f32_bf16(const float* __restrict__ in, unsigned short* __restrict__ out,
             long long n) {
  long long i = (long long)blockIdx.x * blockDim.x + threadIdx.x;
  long long stride = (long long)gridDim.x * blockDim.x;
  for (; i < n; i += stride) out[i] = f32_to_bf16(in[i]);
}

// ---------------------------------------------------------------------------
// Router: sigmoid(x @ rw^T + t_emb @ rtw^T), top-2, normalized gates -> combine[N,8]
// One wave per token.
// ---------------------------------------------------------------------------
__global__ void __launch_bounds__(256)
router_kernel(const float* __restrict__ x, const float* __restrict__ t_emb,
              const float* __restrict__ rw, const float* __restrict__ rtw,
              const float* __restrict__ rbias, float* __restrict__ comb,
              int d, int T) {
  const int lane = threadIdx.x & 31;
  const int wave = threadIdx.x >> 5;
  const int n = blockIdx.x * 8 + wave;
  const int b = n / T;

  const float* xp = x + (size_t)n * d;
  const float* tp = t_emb + (size_t)b * d;

  float acc[8] = {0.f, 0.f, 0.f, 0.f, 0.f, 0.f, 0.f, 0.f};
  for (int i = lane; i < d; i += 32) {
    float xv = xp[i], tv = tp[i];
#pragma unroll
    for (int e = 0; e < 8; ++e)
      acc[e] += xv * rw[(size_t)e * d + i] + tv * rtw[(size_t)e * d + i];
  }
#pragma unroll
  for (int off = 16; off > 0; off >>= 1)
#pragma unroll
    for (int e = 0; e < 8; ++e) acc[e] += __shfl_xor(acc[e], off, 32);

  if (lane == 0) {
    float s[8], sel[8];
#pragma unroll
    for (int e = 0; e < 8; ++e) {
      s[e] = 1.f / (1.f + __expf(-acc[e]));
      sel[e] = s[e] + rbias[e];
    }
    // top-2 with lowest-index tie-break (matches jax.lax.top_k)
    int i1 = 0;
    for (int e = 1; e < 8; ++e) if (sel[e] > sel[i1]) i1 = e;
    int i2 = (i1 == 0) ? 1 : 0;
    for (int e = 0; e < 8; ++e) if (e != i1 && sel[e] > sel[i2]) i2 = e;
    float denom = s[i1] + s[i2];
    float g1, g2;
    if (denom > 1e-9f) {
      g1 = s[i1] / (denom + 1e-9f);
      g2 = s[i2] / (denom + 1e-9f);
    } else {
      g1 = g2 = 0.5f;
    }
#pragma unroll
    for (int e = 0; e < 8; ++e) {
      float c = (e == i1) ? g1 : ((e == i2) ? g2 : 0.f);
      comb[(size_t)n * 8 + e] = c;
    }
  }
}

// ---------------------------------------------------------------------------
// Shared expert up: Hs = silu(x@sw1^T) * (x@sw3^T), bf16  [N, 2048]
// ---------------------------------------------------------------------------
__global__ void __launch_bounds__(256)
moe_shared_up(const unsigned short* __restrict__ xb,
              const unsigned short* __restrict__ w1,
              const unsigned short* __restrict__ w3,
              unsigned short* __restrict__ Hs) {
  const int K = 1024, LDH = 2048;
  const int lane = threadIdx.x & 31;
  const int w = threadIdx.x >> 5;
  const int m0 = blockIdx.x * 128 + (w & 1) * 64;
  const int n0 = blockIdx.y * 128 + (w >> 1) * 32;

  f32x8 acc1[4][2] = {}, acc3[4][2] = {};
  for (int k0 = 0; k0 < K; k0 += 32) {
    bf16x16 a[4], b1[2], b3[2];
#pragma unroll
    for (int mi = 0; mi < 4; ++mi) a[mi] = load_a_frag(xb, K, m0 + mi * 16, k0, lane);
#pragma unroll
    for (int ni = 0; ni < 2; ++ni) {
      b1[ni] = load_b_frag(w1, K, n0 + ni * 16, k0, lane);
      b3[ni] = load_b_frag(w3, K, n0 + ni * 16, k0, lane);
    }
#pragma unroll
    for (int mi = 0; mi < 4; ++mi)
#pragma unroll
      for (int ni = 0; ni < 2; ++ni) {
        acc1[mi][ni] = wmma_bf16(a[mi], b1[ni], acc1[mi][ni]);
        acc3[mi][ni] = wmma_bf16(a[mi], b3[ni], acc3[mi][ni]);
      }
  }
  const int rbase = m0 + ((lane >> 4) << 3);
  const int col_lo = lane & 15;
#pragma unroll
  for (int mi = 0; mi < 4; ++mi)
#pragma unroll
    for (int r = 0; r < 8; ++r) {
      const int row = rbase + mi * 16 + r;
#pragma unroll
      for (int ni = 0; ni < 2; ++ni) {
        const int col = n0 + ni * 16 + col_lo;
        float u = acc1[mi][ni][r];
        float v = (u / (1.f + __expf(-u))) * acc3[mi][ni][r];   // silu(u)*g
        Hs[(size_t)row * LDH + col] = f32_to_bf16(v);
      }
    }
}

// ---------------------------------------------------------------------------
// Routed experts up: Hr[e] = combine[n,e] * gelu(x @ rw1[e]^T), bf16  [E, N, 512]
// Gate folded into the epilogue so the down-projection is one long reduction.
// ---------------------------------------------------------------------------
__global__ void __launch_bounds__(256)
moe_routed_up(const unsigned short* __restrict__ xb,
              const unsigned short* __restrict__ rw1b,
              const float* __restrict__ comb,
              unsigned short* __restrict__ Hr) {
  const int K = 1024, HRD = 512, N = 4096;
  const int e = blockIdx.z;
  const unsigned short* B = rw1b + (size_t)e * HRD * K;
  unsigned short* out = Hr + (size_t)e * N * HRD;

  const int lane = threadIdx.x & 31;
  const int w = threadIdx.x >> 5;
  const int m0 = blockIdx.x * 128 + (w & 1) * 64;
  const int n0 = blockIdx.y * 128 + (w >> 1) * 32;

  f32x8 acc[4][2] = {};
  mma_accum(xb, B, K, m0, n0, lane, acc);

  const int rbase = m0 + ((lane >> 4) << 3);
  const int col_lo = lane & 15;
#pragma unroll
  for (int mi = 0; mi < 4; ++mi)
#pragma unroll
    for (int r = 0; r < 8; ++r) {
      const int row = rbase + mi * 16 + r;
      const float c = comb[(size_t)row * 8 + e];
#pragma unroll
      for (int ni = 0; ni < 2; ++ni) {
        const int col = n0 + ni * 16 + col_lo;
        float u = acc[mi][ni][r];
        float g = 0.5f * u * (1.f + erff(u * 0.70710678118654752f));  // exact gelu
        out[(size_t)row * HRD + col] = f32_to_bf16(g * c);
      }
    }
}

// ---------------------------------------------------------------------------
// Down: out = (Hs @ sw2^T + sum_e Hr[e] @ rw2[e]^T) / 3, fp32  [N, 1024]
// ---------------------------------------------------------------------------
__global__ void __launch_bounds__(256)
moe_down(const unsigned short* __restrict__ Hs,
         const unsigned short* __restrict__ sw2b,
         const unsigned short* __restrict__ Hr,
         const unsigned short* __restrict__ rw2b,
         float* __restrict__ out) {
  const int N = 4096, D = 1024, HS = 2048, HRD = 512;
  const int lane = threadIdx.x & 31;
  const int w = threadIdx.x >> 5;
  const int m0 = blockIdx.x * 128 + (w & 1) * 64;
  const int n0 = blockIdx.y * 128 + (w >> 1) * 32;

  f32x8 acc[4][2] = {};
  mma_accum(Hs, sw2b, HS, m0, n0, lane, acc);
#pragma unroll 1
  for (int e = 0; e < 8; ++e)
    mma_accum(Hr + (size_t)e * N * HRD, rw2b + (size_t)e * D * HRD,
              HRD, m0, n0, lane, acc);

  const float inv = 1.f / 3.f;   // 1 / (N_SHARED + TOP_K)
  const int rbase = m0 + ((lane >> 4) << 3);
  const int col_lo = lane & 15;
#pragma unroll
  for (int mi = 0; mi < 4; ++mi)
#pragma unroll
    for (int r = 0; r < 8; ++r) {
      const int row = rbase + mi * 16 + r;
#pragma unroll
      for (int ni = 0; ni < 2; ++ni) {
        const int col = n0 + ni * 16 + col_lo;
        out[(size_t)row * D + col] = acc[mi][ni][r] * inv;
      }
    }
}

// ---------------------------------------------------------------------------
// Host launcher
// ---------------------------------------------------------------------------
extern "C" void kernel_launch(void* const* d_in, const int* in_sizes, int n_in,
                              void* d_out, int out_size, void* d_ws, size_t ws_size,
                              hipStream_t stream) {
  const float* x     = (const float*)d_in[0];  // [2,2048,1024]
  const float* t_emb = (const float*)d_in[1];  // [2,1024]
  const float* rwt   = (const float*)d_in[2];  // [8,1024]
  const float* rtw   = (const float*)d_in[3];  // [8,1024]
  const float* rbias = (const float*)d_in[4];  // [8]
  const float* sw1   = (const float*)d_in[5];  // [2048,1024]
  const float* sw3   = (const float*)d_in[6];  // [2048,1024]
  const float* sw2   = (const float*)d_in[7];  // [1024,2048]
  const float* rw1   = (const float*)d_in[8];  // [8,512,1024]
  const float* rw2   = (const float*)d_in[9];  // [8,1024,512]
  float* out = (float*)d_out;

  const int N = 4096, D = 1024, HS = 2048, HRD = 512, E = 8, T = 2048;

  char* ws = (char*)d_ws;
  size_t off = 0;
  auto walloc = [&](size_t bytes) -> void* {
    void* p = ws + off;
    off += (bytes + 255) & ~(size_t)255;
    return p;
  };
  unsigned short* xb  = (unsigned short*)walloc((size_t)N * D * 2);
  unsigned short* w1b = (unsigned short*)walloc((size_t)HS * D * 2);
  unsigned short* w3b = (unsigned short*)walloc((size_t)HS * D * 2);
  unsigned short* w2b = (unsigned short*)walloc((size_t)D * HS * 2);
  unsigned short* r1b = (unsigned short*)walloc((size_t)E * HRD * D * 2);
  unsigned short* r2b = (unsigned short*)walloc((size_t)E * D * HRD * 2);
  unsigned short* Hsb = (unsigned short*)walloc((size_t)N * HS * 2);
  unsigned short* Hrb = (unsigned short*)walloc((size_t)E * N * HRD * 2);
  float* comb         = (float*)walloc((size_t)N * E * 4);
  (void)ws_size; (void)in_sizes; (void)n_in; (void)out_size;

  auto cvt = [&](const float* in, unsigned short* o, long long n) {
    int blocks = (int)((n + 256LL * 8 - 1) / (256LL * 8));
    if (blocks > 8192) blocks = 8192;
    cvt_f32_bf16<<<blocks, 256, 0, stream>>>(in, o, n);
  };
  cvt(x,   xb,  (long long)N * D);
  cvt(sw1, w1b, (long long)HS * D);
  cvt(sw3, w3b, (long long)HS * D);
  cvt(sw2, w2b, (long long)D * HS);
  cvt(rw1, r1b, (long long)E * HRD * D);
  cvt(rw2, r2b, (long long)E * D * HRD);

  router_kernel<<<N / 8, 256, 0, stream>>>(x, t_emb, rwt, rtw, rbias, comb, D, T);

  moe_shared_up<<<dim3(N / 128, HS / 128), 256, 0, stream>>>(xb, w1b, w3b, Hsb);
  moe_routed_up<<<dim3(N / 128, HRD / 128, E), 256, 0, stream>>>(xb, r1b, comb, Hrb);
  moe_down<<<dim3(N / 128, D / 128), 256, 0, stream>>>(Hsb, w2b, Hrb, r2b, out);
}
